// SemanticPack_3126736191705
// MI455X (gfx1250) — matrix operand order repaired and checked by reading, not compiled, not run on hardware
//
#include <hip/hip_runtime.h>
#include <hip/hip_bf16.h>

typedef _Float16 v16h __attribute__((ext_vector_type(16)));
typedef _Float16 v8h  __attribute__((ext_vector_type(8)));
typedef float    v8f  __attribute__((ext_vector_type(8)));

#define D_MODEL 1024
#define N_HEADS 16
#define D_HEAD  64
#define TOPK    4
#define SEQ     4096
#define BATCH   16
#define TD      768
#define ID      1280
#define MBANK   32

// ---------------- workspace layout (floats unless noted) ----------------
// Kmat  [16][4][1024] f32      offset 0        (65536)
// Vmat  [16][4][1024] f32      offset 65536    (65536)
// idx   [16][4]       i32      offset 131072   (64)
// sbias [16][64]      f32      offset 131136   (1024)
// QKf   f16 frags [16][32ks][4nt][32lane][16j]  byte offset 528640 (2 MB)
// VWf   f16 frags [16][2ks][64nt][32lane][16j]  byte offset 2625792 (2 MB)
// total ~4.6 MB

// ============ Kernel A1: guidance vector, LN, cosine top-4 ============
__global__ void guide_topk_kernel(const float* __restrict__ text,
                                  const float* __restrict__ image,
                                  const float* __restrict__ mem_keys,
                                  const float* __restrict__ Wt, const float* __restrict__ bt,
                                  const float* __restrict__ Wi, const float* __restrict__ bi,
                                  const float* __restrict__ gng, const float* __restrict__ gnb,
                                  int* __restrict__ idxw) {
    __shared__ float g[D_MODEL];
    __shared__ float red[256];
    __shared__ float red2[256];
    __shared__ float sim[MBANK];
    __shared__ float stat[2];
    const int b = blockIdx.x, tid = threadIdx.x;
    const float* te = text  + (size_t)b * TD;
    const float* im = image + (size_t)b * ID;
    for (int d = tid; d < D_MODEL; d += 256) {
        float acc = bt[d] + bi[d];
        const float* wt = Wt + (size_t)d * TD;
        for (int t = 0; t < TD; ++t) acc += te[t] * wt[t];
        const float* wi = Wi + (size_t)d * ID;
        for (int t = 0; t < ID; ++t) acc += im[t] * wi[t];
        g[d] = acc;
    }
    __syncthreads();
    float s = 0.f, q = 0.f;
    for (int d = tid; d < D_MODEL; d += 256) { float v = g[d]; s += v; q += v * v; }
    red[tid] = s; red2[tid] = q; __syncthreads();
    for (int st = 128; st > 0; st >>= 1) {
        if (tid < st) { red[tid] += red[tid + st]; red2[tid] += red2[tid + st]; }
        __syncthreads();
    }
    if (tid == 0) {
        float mu = red[0] * (1.f / D_MODEL);
        float var = red2[0] * (1.f / D_MODEL) - mu * mu;
        stat[0] = mu; stat[1] = rsqrtf(var + 1e-5f);
    }
    __syncthreads();
    float mu = stat[0], rstd = stat[1];
    s = 0.f;
    for (int d = tid; d < D_MODEL; d += 256) {
        float v = (g[d] - mu) * rstd * gng[d] + gnb[d];
        g[d] = v; s += v * v;
    }
    red[tid] = s; __syncthreads();
    for (int st = 128; st > 0; st >>= 1) {
        if (tid < st) red[tid] += red[tid + st];
        __syncthreads();
    }
    if (tid == 0) stat[0] = 1.f / fmaxf(sqrtf(red[0]), 1e-8f);
    __syncthreads();
    float inv = stat[0];
    if (tid < MBANK) {
        const float* kr = mem_keys + (size_t)tid * D_MODEL;
        float dot = 0.f, nn = 0.f;
        for (int d = 0; d < D_MODEL; ++d) { float kv = kr[d]; dot += g[d] * kv; nn += kv * kv; }
        sim[tid] = dot * inv / fmaxf(sqrtf(nn), 1e-8f);
    }
    __syncthreads();
    if (tid == 0) {
        bool used[MBANK];
        for (int m = 0; m < MBANK; ++m) used[m] = false;
        for (int kk = 0; kk < TOPK; ++kk) {
            int best = 0; float bv = -1e30f;
            for (int m = 0; m < MBANK; ++m)
                if (!used[m] && sim[m] > bv) { bv = sim[m]; best = m; }
            used[best] = true;
            idxw[b * TOPK + kk] = best;
        }
    }
}

// ============ Kernel A2: K/V projection of the 64 gathered tokens ============
// grid = 512, block = 256; each block owns 4 output rows across all (b,k)
__global__ void kv_proj_kernel(const float* __restrict__ mem_values,
                               const float* __restrict__ Wk, const float* __restrict__ bk,
                               const float* __restrict__ Wv, const float* __restrict__ bv,
                               const int* __restrict__ idxw,
                               float* __restrict__ Kmat, float* __restrict__ Vmat) {
    __shared__ float red[256];
    const int tid = threadIdx.x;
    const int tk = tid & 63;      // (b,k) pair 0..63
    const int p  = tid >> 6;      // partial 0..3
    const float* tok = mem_values + (size_t)idxw[tk] * D_MODEL;
    for (int jj = 0; jj < 4; ++jj) {
        const int j = blockIdx.x * 4 + jj;      // 0..2047
        const float* Wrow; const float* bias; float* outm; int jo;
        if (j < D_MODEL) { Wrow = Wk + (size_t)j * D_MODEL;               bias = bk; outm = Kmat; jo = j; }
        else             { Wrow = Wv + (size_t)(j - D_MODEL) * D_MODEL;   bias = bv; outm = Vmat; jo = j - D_MODEL; }
        float acc = 0.f;
        const int i0 = p * 256;
        for (int i = i0; i < i0 + 256; ++i) acc += Wrow[i] * tok[i];
        red[tid] = acc; __syncthreads();
        if (p == 0)
            outm[(size_t)tk * D_MODEL + jo] =
                red[tk] + red[64 + tk] + red[128 + tk] + red[192 + tk] + bias[jo];
        __syncthreads();
    }
}

// ============ Kernel A3: build WMMA B-fragments QKf / VWf + score bias ============
// grid = 16*64 = 1024 blocks, one per (b, hk) with hk = head*4 + token
__global__ void frag_build_kernel(const float* __restrict__ Wq, const float* __restrict__ bq,
                                  const float* __restrict__ Wo,
                                  const float* __restrict__ Kmat, const float* __restrict__ Vmat,
                                  _Float16* __restrict__ QKf, _Float16* __restrict__ VWf,
                                  float* __restrict__ sbias) {
    const int b  = blockIdx.x >> 6;
    const int hk = blockIdx.x & 63;
    const int h  = hk >> 2, kt = hk & 3;
    __shared__ float Kv[D_HEAD], Vv[D_HEAD];
    const int tid = threadIdx.x;
    if (tid < D_HEAD) {
        Kv[tid] = Kmat[((size_t)b * TOPK + kt) * D_MODEL + h * D_HEAD + tid];
        Vv[tid] = Vmat[((size_t)b * TOPK + kt) * D_MODEL + h * D_HEAD + tid];
    }
    __syncthreads();
    // QKmat column hk:  QK[kg, hk] = sum_d Wq[h*64+d, kg] * Kv[d]
    for (int kg = tid; kg < D_MODEL; kg += 256) {
        float acc = 0.f;
        #pragma unroll 8
        for (int d = 0; d < D_HEAD; ++d) acc += Wq[(size_t)(h * D_HEAD + d) * D_MODEL + kg] * Kv[d];
        const int ks = kg >> 5, rem = kg & 31, lh = (rem >> 3) & 1, j = (rem & 7) | ((rem >> 4) << 3);
        const int lane = lh * 16 + (hk & 15), nt = hk >> 4;
        QKf[((((size_t)b * 32 + ks) * 4 + nt) * 32 + lane) * 16 + j] = (_Float16)acc;
    }
    // VW row hk: VW[hk, jg] = sum_d Vv[d] * Wo[jg, h*64+d]
    for (int jg = tid; jg < D_MODEL; jg += 256) {
        float acc = 0.f;
        #pragma unroll 8
        for (int d = 0; d < D_HEAD; ++d) acc += Vv[d] * Wo[(size_t)jg * D_MODEL + h * D_HEAD + d];
        const int ks2 = hk >> 5, lh = (hk >> 3) & 1, j = (hk & 7) | (((hk & 31) >> 4) << 3);
        const int nt2 = jg >> 4, lane = lh * 16 + (jg & 15);
        VWf[((((size_t)b * 2 + ks2) * 64 + nt2) * 32 + lane) * 16 + j] = (_Float16)acc;
    }
    if (tid == 0) {
        float sacc = 0.f;
        for (int d = 0; d < D_HEAD; ++d) sacc += bq[h * D_HEAD + d] * Kv[d];
        sbias[b * 64 + hk] = sacc;
    }
}

// ============ Kernel B: fused streaming scores->softmax->out->residual->LN ============
// grid = 16*64 = 1024 blocks, block = 128 threads (4 waves), wave owns 16 rows.
// per-wave dynamic LDS: tmp[16][1028] f32 (65792B) + attn[16][64] f16 (2048B) + stats[32] f32 (128B)
#define WAVE_LDS 67968
#define TSTRIDE  1028

__global__ __launch_bounds__(128, 1)
void fused_attn_ln_kernel(const float* __restrict__ x,
                          const _Float16* __restrict__ QKf,
                          const _Float16* __restrict__ VWf,
                          const float* __restrict__ sbias,
                          const float* __restrict__ bo,
                          const float* __restrict__ lng, const float* __restrict__ lnb,
                          float* __restrict__ out) {
    extern __shared__ char smem[];
    const int tid  = threadIdx.x;
    const int wave = tid >> 5;
    const int lane = tid & 31;
    const int half = lane >> 4;
    const int n15  = lane & 15;
    const int b    = blockIdx.x >> 6;
    const size_t rowbase = (size_t)b * SEQ + (size_t)(blockIdx.x & 63) * 64 + wave * 16;

    float*    tmpf  = (float*)   (smem + (size_t)wave * WAVE_LDS);
    _Float16* attnh = (_Float16*)(smem + (size_t)wave * WAVE_LDS + 65792);
    float*    stats = (float*)   (smem + (size_t)wave * WAVE_LDS + 65792 + 2048);

    const v8f z = {0.f, 0.f, 0.f, 0.f, 0.f, 0.f, 0.f, 0.f};

    // ---- Phase 1: scores GEMM  [16 x 1024] x [1024 x 64]  (f16 WMMA, f32 acc) ----
    v8f accS[4]; accS[0] = z; accS[1] = z; accS[2] = z; accS[3] = z;
    const float*    xrow = x + (rowbase + n15) * D_MODEL;
    const _Float16* qk   = QKf + (size_t)b * 32 * 4 * 32 * 16 + lane * 16;
    #pragma unroll 4
    for (int ks = 0; ks < 32; ++ks) {
        const float* xp = xrow + ks * 32 + half * 8;
        // keep the x stream and the fragment stream warm ~8 iterations ahead
        __builtin_prefetch(xp + 256, 0, 0);                       // global_prefetch_b8
        __builtin_prefetch(qk + (size_t)ks * 2048 + 16384, 0, 0); // QKf, 8 ks ahead
        float4 f0 = *(const float4*)(xp);
        float4 f1 = *(const float4*)(xp + 4);
        float4 f2 = *(const float4*)(xp + 16);
        float4 f3 = *(const float4*)(xp + 20);
        // stash residual tile in LDS (x is read exactly once from HBM)
        float* tr = tmpf + n15 * TSTRIDE + ks * 32 + half * 8;
        *(float4*)tr        = f0;  *(float4*)(tr + 4)  = f1;
        *(float4*)(tr + 16) = f2;  *(float4*)(tr + 20) = f3;
        v16h a;
        a[0]  = (_Float16)f0.x; a[1]  = (_Float16)f0.y; a[2]  = (_Float16)f0.z; a[3]  = (_Float16)f0.w;
        a[4]  = (_Float16)f1.x; a[5]  = (_Float16)f1.y; a[6]  = (_Float16)f1.z; a[7]  = (_Float16)f1.w;
        a[8]  = (_Float16)f2.x; a[9]  = (_Float16)f2.y; a[10] = (_Float16)f2.z; a[11] = (_Float16)f2.w;
        a[12] = (_Float16)f3.x; a[13] = (_Float16)f3.y; a[14] = (_Float16)f3.z; a[15] = (_Float16)f3.w;
        const _Float16* bp = qk + (size_t)ks * 2048;
        accS[0] = __builtin_amdgcn_wmma_f32_16x16x32_f16(false, a, false, *(const v16h*)(bp),        (short)0, accS[0], false, false);
        accS[1] = __builtin_amdgcn_wmma_f32_16x16x32_f16(false, a, false, *(const v16h*)(bp + 512),  (short)0, accS[1], false, false);
        accS[2] = __builtin_amdgcn_wmma_f32_16x16x32_f16(false, a, false, *(const v16h*)(bp + 1024), (short)0, accS[2], false, false);
        accS[3] = __builtin_amdgcn_wmma_f32_16x16x32_f16(false, a, false, *(const v16h*)(bp + 1536), (short)0, accS[3], false, false);
    }

    // ---- Phase 2: per-head softmax over 4 tokens (groups of 4 lanes) ----
    const _Float16* vw = VWf + (size_t)b * 2 * 64 * 512 + lane * 16;
    const float scale = 0.125f; // 1/sqrt(64)
    #pragma unroll
    for (int nt = 0; nt < 4; ++nt) {
        // warm the VWf region while doing the softmax math
        __builtin_prefetch(vw + (size_t)nt * 16384, 0, 0);
        const float sb = sbias[b * 64 + nt * 16 + n15];
        #pragma unroll
        for (int i = 0; i < 8; ++i) {
            float sv = accS[nt][i] * scale + sb;
            float m1 = fmaxf(sv, __shfl_xor(sv, 1));
            float m2 = fmaxf(m1, __shfl_xor(m1, 2));
            float e  = __expf(sv - m2);
            float t1 = e + __shfl_xor(e, 1);
            float t2 = t1 + __shfl_xor(t1, 2);
            attnh[(i + half * 8) * 64 + nt * 16 + n15] = (_Float16)(e / t2);
        }
    }
    __syncthreads();

    // ---- Phase 3: out GEMM [16 x 64] x [64 x 1024] + residual + bo + LN stats ----
    const _Float16* ar = attnh + n15 * 64;
    v8h lo0 = *(const v8h*)(ar + half * 8);
    v8h hi0 = *(const v8h*)(ar + 16 + half * 8);
    v8h lo1 = *(const v8h*)(ar + 32 + half * 8);
    v8h hi1 = *(const v8h*)(ar + 48 + half * 8);
    v16h a0, a1;
    #pragma unroll
    for (int j = 0; j < 8; ++j) {
        a0[j] = lo0[j]; a0[j + 8] = hi0[j];
        a1[j] = lo1[j]; a1[j + 8] = hi1[j];
    }
    float rs[8], rq[8];
    #pragma unroll
    for (int i = 0; i < 8; ++i) { rs[i] = 0.f; rq[i] = 0.f; }
    #pragma unroll 8
    for (int nt2 = 0; nt2 < 64; ++nt2) {
        __builtin_prefetch(vw + (size_t)(nt2 + 8) * 512, 0, 0);
        __builtin_prefetch(vw + (size_t)(72 + nt2) * 512, 0, 0);
        v8f acc = z;
        acc = __builtin_amdgcn_wmma_f32_16x16x32_f16(false, a0, false, *(const v16h*)(vw + (size_t)nt2 * 512),        (short)0, acc, false, false);
        acc = __builtin_amdgcn_wmma_f32_16x16x32_f16(false, a1, false, *(const v16h*)(vw + (size_t)(64 + nt2) * 512), (short)0, acc, false, false);
        const int col = nt2 * 16 + n15;
        const float bov = bo[col];
        #pragma unroll
        for (int i = 0; i < 8; ++i) {
            const int row = i + half * 8;
            float t = tmpf[row * TSTRIDE + col] + acc[i] + bov;
            tmpf[row * TSTRIDE + col] = t;
            rs[i] += t; rq[i] += t * t;
        }
    }
    #pragma unroll
    for (int m = 1; m < 16; m <<= 1) {
        #pragma unroll
        for (int i = 0; i < 8; ++i) {
            rs[i] += __shfl_xor(rs[i], m);
            rq[i] += __shfl_xor(rq[i], m);
        }
    }
    if (n15 == 0) {
        #pragma unroll
        for (int i = 0; i < 8; ++i) {
            float mean = rs[i] * (1.f / D_MODEL);
            float var  = rq[i] * (1.f / D_MODEL) - mean * mean;
            stats[(half * 8 + i) * 2]     = mean;
            stats[(half * 8 + i) * 2 + 1] = rsqrtf(var + 1e-5f);
        }
    }
    __syncthreads();

    // ---- Phase 4: apply LayerNorm, coalesced float4 stores ----
    float* op = out + rowbase * D_MODEL;
    for (int row = 0; row < 16; ++row) {
        const float mean = stats[row * 2], rstd = stats[row * 2 + 1];
        const float* trow = tmpf + row * TSTRIDE;
        float* orow = op + (size_t)row * D_MODEL;
        #pragma unroll
        for (int c0 = 0; c0 < 8; ++c0) {
            const int col = c0 * 128 + lane * 4;
            float4 t  = *(const float4*)(trow + col);
            float4 g4 = *(const float4*)(lng + col);
            float4 b4 = *(const float4*)(lnb + col);
            float4 r;
            r.x = (t.x - mean) * rstd * g4.x + b4.x;
            r.y = (t.y - mean) * rstd * g4.y + b4.y;
            r.z = (t.z - mean) * rstd * g4.z + b4.z;
            r.w = (t.w - mean) * rstd * g4.w + b4.w;
            *(float4*)(orow + col) = r;
        }
    }
}

// ============================== launch ==============================
extern "C" void kernel_launch(void* const* d_in, const int* in_sizes, int n_in,
                              void* d_out, int out_size, void* d_ws, size_t ws_size,
                              hipStream_t stream) {
    const float* x        = (const float*)d_in[0];
    const float* text_emb = (const float*)d_in[1];
    const float* image_emb= (const float*)d_in[2];
    const float* mem_keys = (const float*)d_in[3];
    const float* mem_vals = (const float*)d_in[4];
    const float* Wt  = (const float*)d_in[5];
    const float* bt  = (const float*)d_in[6];
    const float* Wi  = (const float*)d_in[7];
    const float* bi  = (const float*)d_in[8];
    const float* gng = (const float*)d_in[9];
    const float* gnb = (const float*)d_in[10];
    const float* Wq  = (const float*)d_in[11];
    const float* bq  = (const float*)d_in[12];
    const float* Wk  = (const float*)d_in[13];
    const float* bk  = (const float*)d_in[14];
    const float* Wv  = (const float*)d_in[15];
    const float* bv  = (const float*)d_in[16];
    const float* Wo  = (const float*)d_in[17];
    const float* bo  = (const float*)d_in[18];
    const float* lng = (const float*)d_in[19];
    const float* lnb = (const float*)d_in[20];

    float* ws = (float*)d_ws;
    float*    Kmat  = ws;
    float*    Vmat  = ws + 65536;
    int*      idxw  = (int*)(ws + 131072);
    float*    sbias = ws + 131136;
    _Float16* QKf   = (_Float16*)((char*)d_ws + 528640);
    _Float16* VWf   = (_Float16*)((char*)d_ws + 2625792);

    guide_topk_kernel<<<BATCH, 256, 0, stream>>>(text_emb, image_emb, mem_keys,
                                                 Wt, bt, Wi, bi, gng, gnb, idxw);
    kv_proj_kernel<<<512, 256, 0, stream>>>(mem_vals, Wk, bk, Wv, bv, idxw, Kmat, Vmat);
    frag_build_kernel<<<1024, 256, 0, stream>>>(Wq, bq, Wo, Kmat, Vmat, QKf, VWf, sbias);

    const size_t shmem = (size_t)4 * WAVE_LDS; // 271872 bytes (<= 320KB WGP LDS)
    hipFuncSetAttribute(reinterpret_cast<const void*>(fused_attn_ln_kernel),
                        hipFuncAttributeMaxDynamicSharedMemorySize, (int)shmem);
    fused_attn_ln_kernel<<<1024, 128, shmem, stream>>>(x, QKf, VWf, sbias, bo, lng, lnb,
                                                       (float*)d_out);
}
